// CNN_VQC_32272384262283
// MI455X (gfx1250) — matrix operand order repaired and checked by reading, not compiled, not run on hardware
//
#include <hip/hip_runtime.h>
#include <hip/hip_bf16.h>
#include <math.h>

typedef __attribute__((ext_vector_type(16))) _Float16 v16h;
typedef __attribute__((ext_vector_type(8)))  float    v8f;
typedef __attribute__((ext_vector_type(4)))  unsigned u32x4;
typedef __attribute__((ext_vector_type(8)))  unsigned u32x8;

#define BATCH 64
#define NQ 16
#define NL 6
#define DIM 65536          // 2^16 amplitudes
#define TILE_SZ 8192       // 64 KB of float2 in LDS
#define PI_F 3.14159265358979f

// -------------------------------------------------------------------------
// init: zero pooled-feature accumulator, set |0...0> for each sample
// -------------------------------------------------------------------------
__global__ void k_init(float* __restrict__ feat, float2* __restrict__ state) {
  int gid = blockIdx.x * blockDim.x + threadIdx.x;
  if (gid < BATCH * DIM)
    state[gid] = make_float2(((gid & (DIM - 1)) == 0) ? 1.f : 0.f, 0.f);
  if (gid < BATCH * 512) feat[gid] = 0.f;
}

// -------------------------------------------------------------------------
// Fused conv(4x4,s4) + bias + ReLU + global-avg-pool (sum; scaled later)
// Implicit GEMM, WMMA f16 16x16x32, K = 48 zero-padded to 64 (2 WMMAs).
// Grid: (28 pixel-groups of 112 px, 4 channel-groups of 128, 64 batch)
// Block: 256 threads = 8 waves; each wave owns a 16-channel slice.
// -------------------------------------------------------------------------
__global__ void __launch_bounds__(256)
k_conv_pool(const float* __restrict__ x, const float* __restrict__ Wc,
            const float* __restrict__ bc, float* __restrict__ feat) {
  __shared__ _Float16 Bt[128][64];   // [channel][k] weight tile (f16)
  __shared__ _Float16 At[16][64];    // [pixel][k]   im2col tile  (f16)

  const int tid    = threadIdx.x;
  const int b      = blockIdx.z;
  const int chbase = blockIdx.y * 128;
  const int wv     = tid >> 5;
  const int lane   = tid & 31;
  const int g      = lane >> 4;      // lane-half selects K sub-block
  const int nn     = lane & 15;      // M row (A) / N col (B,C)

  // weight tile: 128 channels x 48 K (padded to 64), loaded once
  for (int idx = tid; idx < 128 * 64; idx += 256) {
    int ch = idx >> 6, k = idx & 63;
    Bt[ch][k] = (_Float16)((k < 48) ? Wc[(chbase + ch) * 48 + k] : 0.f);
  }

  const float bias = bc[chbase + wv * 16 + nn];
  float acc = 0.f;

  for (int t = 0; t < 7; ++t) {
    const int pbase = (blockIdx.x * 7 + t) * 16;
    __syncthreads();                 // guards Bt (t=0) and At reuse (t>0)

    // im2col: 16 output pixels x 48 K values, f32 -> f16
    for (int idx = tid; idx < 16 * 64; idx += 256) {
      int m = idx >> 6, k = idx & 63;
      float v = 0.f;
      if (k < 48) {
        int c = k >> 4, ky = (k >> 2) & 3, kx = k & 3;
        int p = pbase + m;
        if (p < 3136) {
          int oy = p / 56, ox = p % 56;
          v = x[((size_t)(b * 3 + c) * 224 + oy * 4 + ky) * 224 + ox * 4 + kx];
        }
      }
      At[m][k] = (_Float16)v;
    }
    // prefetch next pixel tile's input region into cache
    if (t < 6) {
      int pn = pbase + 16;
      __builtin_prefetch(
          &x[((size_t)(b * 3) * 224 + (pn / 56) * 4) * 224 + (pn % 56) * 4 + (tid & 63)],
          0, 1);
    }
    __syncthreads();

    v8f cacc = {};
    const unsigned* arow = (const unsigned*)&At[nn][0];
    const unsigned* brow = (const unsigned*)&Bt[wv * 16 + nn][0];
#pragma unroll
    for (int s = 0; s < 2; ++s) {
      const int kbase = s * 32;
      union { unsigned u[8]; v16h h; } fa, fb;
#pragma unroll
      for (int v = 0; v < 8; ++v) {
        int kla = ((v < 4) ? 0 : 16) + g * 8 + (v & 3) * 2;  // A 16x32 layout
        fa.u[v] = arow[(kbase + kla) >> 1];
        int klb = ((v < 4) ? 0 : 16) + g * 8 + (v & 3) * 2;  // B 32x16 layout
        fb.u[v] = brow[(kbase + klb) >> 1];
      }
      cacc = __builtin_amdgcn_wmma_f32_16x16x32_f16(
          false, fa.h, false, fb.h, (short)0, cacc, false, false);
    }
    // bias + ReLU + partial pool over the 8 M rows this lane holds
#pragma unroll
    for (int v = 0; v < 8; ++v) {
      float val = cacc[v] + bias;
      acc += (val > 0.f) ? val : 0.f;
    }
  }
  // combine the two M-halves (lane L and L^16 share channel nn)
  float full = acc + __shfl_xor(acc, 16, 32);
  if (lane < 16)
    atomicAdd(&feat[b * 512 + chbase + wv * 16 + nn], full);
}

// -------------------------------------------------------------------------
// Reducer: angles = pi * tanh( (feat/3136) @ W_red^T + b_red )
// GEMM M=64,N=16,K=512 via WMMA: 1 workgroup, 4 waves = 4 M-tiles.
// -------------------------------------------------------------------------
__global__ void __launch_bounds__(128)
k_reduce(const float* __restrict__ feat, const float* __restrict__ Wr,
         const float* __restrict__ br, float* __restrict__ angles) {
  const int tid  = threadIdx.x;
  const int wv   = tid >> 5;
  const int lane = tid & 31;
  const int g    = lane >> 4;
  const int nn   = lane & 15;
  const float scale = 1.f / 3136.f;

  v8f cacc = {};
  for (int kb = 0; kb < 512; kb += 32) {
    union { _Float16 q[16]; v16h h; } fa, fb;
#pragma unroll
    for (int v = 0; v < 8; ++v) {
      int kla = ((v < 4) ? 0 : 16) + g * 8 + (v & 3) * 2;
      int ka  = kb + kla;
      int smp = wv * 16 + nn;
      fa.q[2 * v]     = (_Float16)(feat[smp * 512 + ka] * scale);
      fa.q[2 * v + 1] = (_Float16)(feat[smp * 512 + ka + 1] * scale);
      int kbf = kb + ((v < 4) ? 0 : 16) + g * 8 + (v & 3) * 2; // B[k][n] = W_red[n][k]
      fb.q[2 * v]     = (_Float16)Wr[nn * 512 + kbf];
      fb.q[2 * v + 1] = (_Float16)Wr[nn * 512 + kbf + 1];
    }
    cacc = __builtin_amdgcn_wmma_f32_16x16x32_f16(
        false, fa.h, false, fb.h, (short)0, cacc, false, false);
  }
  const float bias = br[nn];
#pragma unroll
  for (int v = 0; v < 8; ++v) {
    int m   = v + 8 * g;
    int smp = wv * 16 + m;
    angles[smp * 16 + nn] = tanhf(cacc[v] + bias) * PI_F;
  }
}

// -------------------------------------------------------------------------
// VQC gate primitives on an LDS-resident tile of 8192 amplitudes.
// wire w lives at global bit (15-w); p below is the *local* bit position.
// -------------------------------------------------------------------------
static __device__ __forceinline__ void ry_pairs(float2* S, int tid, int p,
                                                float c, float s) {
  for (int j = 0; j < TILE_SZ / 2 / 256; ++j) {
    int idx = j * 256 + tid;
    unsigned mask = (1u << p) - 1u;
    unsigned i0 = ((unsigned)(idx >> p) << (p + 1)) | ((unsigned)idx & mask);
    unsigned i1 = i0 | (1u << p);
    float2 a = S[i0], b = S[i1];
    S[i0] = make_float2(c * a.x - s * b.x, c * a.y - s * b.y);
    S[i1] = make_float2(s * a.x + c * b.x, s * a.y + c * b.y);
  }
  __syncthreads();
}
static __device__ __forceinline__ void rz_all(float2* S, int tid, int p,
                                              float c, float s) {
  for (int j = 0; j < TILE_SZ / 256; ++j) {
    int l = j * 256 + tid;
    float sg = ((l >> p) & 1) ? s : -s;   // e^{-i t/2} / e^{+i t/2}
    float2 a = S[l];
    S[l] = make_float2(c * a.x - sg * a.y, c * a.y + sg * a.x);
  }
  __syncthreads();
}
static __device__ __forceinline__ void cnot_pairs(float2* S, int tid, int pc,
                                                  int pt) {
  for (int j = 0; j < TILE_SZ / 2 / 256; ++j) {
    int idx = j * 256 + tid;
    unsigned mask = (1u << pt) - 1u;
    unsigned i0 = ((unsigned)(idx >> pt) << (pt + 1)) | ((unsigned)idx & mask);
    if ((i0 >> pc) & 1u) {
      unsigned i1 = i0 | (1u << pt);
      float2 a = S[i0]; S[i0] = S[i1]; S[i1] = a;
    }
  }
  __syncthreads();
}

// -------------------------------------------------------------------------
// Low pass: tile = {global bits 15..12} x {bits 8..0}; tile coord = bits 11..9
// Handles wires 0..3 (local p = 12-w) and CNOT(0,1)(1,2)(2,3).
// float4 traffic: 2 amplitudes per thread per access.
// -------------------------------------------------------------------------
__global__ void __launch_bounds__(256)
k_vqc_low(float2* __restrict__ state, const float* __restrict__ angles,
          const float* __restrict__ wts, int layer) {
  __shared__ float2 S[TILE_SZ];
  const int tid = threadIdx.x;
  const unsigned t = blockIdx.x;           // 3 bits -> global bits 11..9
  const int b = blockIdx.y;
  float2* st = state + (size_t)b * DIM;

  for (int j = 0; j < TILE_SZ / 512; ++j) {
    int l = j * 512 + tid * 2;             // pair of consecutive amplitudes
    unsigned i = ((unsigned)(l >> 9) << 12) | (t << 9) | (unsigned)(l & 511);
    *(float4*)&S[l] = *(const float4*)&st[i];
  }
  __syncthreads();

  if (layer < 0) {                          // encoding: RY(angles) wires 0..3
    for (int w = 0; w < 4; ++w) {
      float th = angles[b * 16 + w] * 0.5f;
      ry_pairs(S, tid, 12 - w, cosf(th), sinf(th));
    }
  } else {
    for (int w = 0; w < 4; ++w) {
      float ty = wts[(layer * 16 + w) * 2 + 0] * 0.5f;
      ry_pairs(S, tid, 12 - w, cosf(ty), sinf(ty));
      float tz = wts[(layer * 16 + w) * 2 + 1] * 0.5f;
      rz_all(S, tid, 12 - w, cosf(tz), sinf(tz));
    }
    for (int c = 0; c < 3; ++c) cnot_pairs(S, tid, 12 - c, 11 - c);
  }

  for (int j = 0; j < TILE_SZ / 512; ++j) {
    int l = j * 512 + tid * 2;
    unsigned i = ((unsigned)(l >> 9) << 12) | (t << 9) | (unsigned)(l & 511);
    *(float4*)&st[i] = *(const float4*)&S[l];
  }
}

// -------------------------------------------------------------------------
// High pass: contiguous tile; local bits = global bits 12..0.
// Handles wires 4..15 (local p = 15-w) and CNOT(3,4)..(14,15).
// Tile movement via the Tensor Data Mover (64 KB 1-D DMA, TENSORcnt).
// -------------------------------------------------------------------------
static __device__ __forceinline__ u32x4 tdm_group0(unsigned lds_off,
                                                   unsigned long long ga) {
  u32x4 g0;
  g0[0] = 0x1u;                                    // count=1, no gather
  g0[1] = lds_off;                                 // lds_addr (bytes)
  g0[2] = (unsigned)(ga & 0xffffffffu);            // global_addr[31:0]
  g0[3] = (unsigned)((ga >> 32) & 0x01ffffffu)     // global_addr[56:32]
        | 0x80000000u;                             // type=2 (image)
  return g0;
}
static __device__ __forceinline__ u32x8 tdm_group1() {
  u32x8 g1;
  g1[0] = 0x3u << 16;     // workgroup_mask=0, data_size=3 (8 bytes)
  g1[1] = 0;              // tensor_dim0[15:0] = 0 (65536 & 0xffff)
  g1[2] = 0x00010001u;    // tensor_dim0[31:16]=1 (=65536), tensor_dim1 lo=1
  g1[3] = 0x20000000u;    // tensor_dim1 hi=0, tile_dim0 = 8192
  g1[4] = 0x1u;           // tile_dim1 = 1, tile_dim2 = 0
  g1[5] = 0x00010000u;    // tensor_dim0_stride = 65536 (8B units)
  g1[6] = 0;              // stride hi, tensor_dim1_stride lo
  g1[7] = 0;              // tensor_dim1_stride hi
  return g1;
}

__global__ void __launch_bounds__(256)
k_vqc_high(float2* __restrict__ state, const float* __restrict__ angles,
           const float* __restrict__ wts, int layer) {
  __shared__ float2 S[TILE_SZ];
  const int tid = threadIdx.x;
  const unsigned t = blockIdx.x;           // 3 bits -> global bits 15..13
  const int b = blockIdx.y;
  float2* st = state + (size_t)b * DIM;
  float2* gtile = st + ((size_t)t << 13);  // contiguous 64 KB tile

  const unsigned lds_off = (unsigned)(size_t)&S[0];      // low 32 = LDS offset
  const unsigned long long ga = (unsigned long long)(size_t)gtile;

  // ---- TDM: global -> LDS, issued once by wave 0, tracked via TENSORcnt ----
  if ((tid >> 5) == 0) {
    u32x4 g0 = tdm_group0(lds_off, ga);
    u32x8 g1 = tdm_group1();
    asm volatile("tensor_load_to_lds %0, %1" :: "s"(g0), "s"(g1) : "memory");
    __builtin_amdgcn_s_wait_tensorcnt(0);
  }
  __syncthreads();                          // publish tile to all 8 waves

  if (layer < 0) {                          // encoding: RY(angles) wires 4..15
    for (int w = 4; w < NQ; ++w) {
      float th = angles[b * 16 + w] * 0.5f;
      ry_pairs(S, tid, 15 - w, cosf(th), sinf(th));
    }
  } else {
    for (int w = 4; w < NQ; ++w) {
      float ty = wts[(layer * 16 + w) * 2 + 0] * 0.5f;
      ry_pairs(S, tid, 15 - w, cosf(ty), sinf(ty));
      float tz = wts[(layer * 16 + w) * 2 + 1] * 0.5f;
      rz_all(S, tid, 15 - w, cosf(tz), sinf(tz));
    }
    for (int c = 3; c < NQ - 1; ++c) cnot_pairs(S, tid, 15 - c, 14 - c);
  }
  // gate helpers end with __syncthreads(): LDS writes complete & visible

  // ---- TDM: LDS -> global ----
  if ((tid >> 5) == 0) {
    u32x4 g0 = tdm_group0(lds_off, ga);
    u32x8 g1 = tdm_group1();
    asm volatile("tensor_store_from_lds %0, %1" :: "s"(g0), "s"(g1) : "memory");
    __builtin_amdgcn_s_wait_tensorcnt(0);
  }
}

// -------------------------------------------------------------------------
// <Z_0> then sigmoid; wire 0 = bit 15 (pair of consecutive amps shares sign)
// -------------------------------------------------------------------------
__global__ void __launch_bounds__(256)
k_expect(const float2* __restrict__ state, float* __restrict__ out) {
  __shared__ float red[256];
  const int tid = threadIdx.x, b = blockIdx.x;
  const float2* st = state + (size_t)b * DIM;
  float acc = 0.f;
  for (int j = 0; j < DIM / 512; ++j) {
    int i = j * 512 + tid * 2;
    float4 a = *(const float4*)&st[i];
    float p = a.x * a.x + a.y * a.y + a.z * a.z + a.w * a.w;
    acc += (i & 0x8000) ? -p : p;
  }
  red[tid] = acc;
  __syncthreads();
  for (int s = 128; s > 0; s >>= 1) {
    if (tid < s) red[tid] += red[tid + s];
    __syncthreads();
  }
  if (tid == 0) out[b] = 1.f / (1.f + expf(-red[0]));
}

// -------------------------------------------------------------------------
extern "C" void kernel_launch(void* const* d_in, const int* in_sizes, int n_in,
                              void* d_out, int out_size, void* d_ws, size_t ws_size,
                              hipStream_t stream) {
  const float* x  = (const float*)d_in[0];   // [64,3,224,224]
  const float* Wc = (const float*)d_in[1];   // [512,3,4,4]
  const float* bc = (const float*)d_in[2];   // [512]
  const float* Wr = (const float*)d_in[3];   // [16,512]
  const float* br = (const float*)d_in[4];   // [16]
  const float* vw = (const float*)d_in[5];   // [6,16,2]
  float* out = (float*)d_out;

  float*  feat   = (float*)d_ws;                        // 64*512
  float*  angles = feat + BATCH * 512;                  // 64*16
  float2* state  = (float2*)(feat + BATCH * 512 + 1024);// 64*65536 complex

  k_init<<<(BATCH * DIM + 255) / 256, 256, 0, stream>>>(feat, state);
  k_conv_pool<<<dim3(28, 4, BATCH), 256, 0, stream>>>(x, Wc, bc, feat);
  k_reduce<<<1, 128, 0, stream>>>(feat, Wr, br, angles);

  k_vqc_low <<<dim3(8, BATCH), 256, 0, stream>>>(state, angles, vw, -1);
  k_vqc_high<<<dim3(8, BATCH), 256, 0, stream>>>(state, angles, vw, -1);
  for (int l = 0; l < NL; ++l) {
    k_vqc_low <<<dim3(8, BATCH), 256, 0, stream>>>(state, angles, vw, l);
    k_vqc_high<<<dim3(8, BATCH), 256, 0, stream>>>(state, angles, vw, l);
  }
  k_expect<<<BATCH, 256, 0, stream>>>(state, out);
}
